// TASFTAttention_73306501808593
// MI455X (gfx1250) — compile-verified, matrix operand order checked by reading, not run on hardware
//
#include <hip/hip_runtime.h>
#include <hip/hip_bf16.h>

// ---------------------------------------------------------------------------
// Problem constants (from reference)
// ---------------------------------------------------------------------------
constexpr int SEQ = 2048;   // S
constexpr int EMB = 4096;   // E
constexpr int NH  = 32;     // heads
constexpr int HD  = 128;    // head dim D
constexpr int BLK = 64;     // block size BS
constexpr int NBK = 32;     // SEQ / BLK
constexpr int GD  = 64;     // gate dim

typedef __attribute__((ext_vector_type(16))) __bf16 v16bf;
typedef __attribute__((ext_vector_type(8)))  __bf16 v8bf;
typedef __attribute__((ext_vector_type(8)))  float  v8f;

// ---------------------------------------------------------------------------
// WMMA helpers
// ---------------------------------------------------------------------------
__device__ __forceinline__ v8f wmma_bf16(v16bf a, v16bf b, v8f c) {
    // D = A(16x32) * B(32x16) + C, f32 accumulate
    return __builtin_amdgcn_wmma_f32_16x16x32_bf16(
        /*neg_a=*/false, a, /*neg_b=*/false, b,
        /*c_mod=*/(short)0, c, /*reuse_a=*/false, /*reuse_b=*/false);
}

// Load a 16x32 bf16 operand tile from a row-major [rows x ld] array.
// Per ISA 7.12.2 (16-bit operands): lane L holds row = L%16, K chunks
// [base, base+7] and [base+16, base+23] with base = 8*(L/16).
__device__ __forceinline__ v16bf load_tile_bf16(const __bf16* p, int ld, int lane) {
    const __bf16* r = p + (size_t)(lane & 15) * ld + ((lane >> 4) << 3);
    v8bf lo = *(const v8bf*)(r);
    v8bf hi = *(const v8bf*)(r + 16);
    return __builtin_shufflevector(lo, hi, 0, 1, 2, 3, 4, 5, 6, 7,
                                           8, 9, 10, 11, 12, 13, 14, 15);
}

// ---------------------------------------------------------------------------
// CDNA5 async global->LDS copy (ASYNCcnt path).
// LDS byte address = low 32 bits of the generic pointer (ISA 10.2: LDS
// aperture address truncates to LDS_ADDR.U32).
// ---------------------------------------------------------------------------
__device__ __forceinline__ void async_copy_b128(void* lds_ptr, const void* gptr) {
    unsigned lds = (unsigned)(size_t)lds_ptr;
    asm volatile("global_load_async_to_lds_b128 %0, %1, off"
                 :: "v"(lds), "v"(gptr)
                 : "memory");
}
__device__ __forceinline__ void wait_asynccnt0() {
    asm volatile("s_wait_asynccnt 0x0" ::: "memory");
}

// ---------------------------------------------------------------------------
// f32 -> bf16 conversion
// ---------------------------------------------------------------------------
__global__ void __launch_bounds__(256) k_f32_to_bf16(const float* __restrict__ src,
                                                     __bf16* __restrict__ dst, int n) {
    int i = blockIdx.x * 256 + threadIdx.x;
    if (i < n) dst[i] = (__bf16)src[i];
}

// ---------------------------------------------------------------------------
// GEMM:  C[M x N] = A[M x K] * B[N x K]^T   (both operands row-major over K)
// 256 threads = 8 waves in a 2x4 grid; per-wave register tile 32(M) x 64(N):
// 2 A-tiles + 4 B-tiles -> 8 WMMAs per 32-wide k-step.
// Macro tile per block: 64(M) x 256(N).
// ---------------------------------------------------------------------------
__global__ void __launch_bounds__(256) k_gemm_bf16_nt(const __bf16* __restrict__ A,
                                                      const __bf16* __restrict__ B,
                                                      float* __restrict__ C,
                                                      int M, int N, int K) {
    const int lane = threadIdx.x & 31;
    const int w    = threadIdx.x >> 5;      // 0..7
    const int mw   = w >> 2;                // 0..1
    const int nw   = w & 3;                 // 0..3

    const int row0 = blockIdx.x * 64 + mw * 32;
    const int col0 = blockIdx.y * 256 + nw * 64;

    const __bf16* a0p = A + (size_t)row0 * K;
    const __bf16* a1p = A + (size_t)(row0 + 16) * K;
    const __bf16* b0p = B + (size_t)col0 * K;
    const __bf16* b1p = B + (size_t)(col0 + 16) * K;
    const __bf16* b2p = B + (size_t)(col0 + 32) * K;
    const __bf16* b3p = B + (size_t)(col0 + 48) * K;

    v8f acc[2][4] = {};
    for (int k = 0; k < K; k += 32) {
        v16bf a0 = load_tile_bf16(a0p + k, K, lane);
        v16bf a1 = load_tile_bf16(a1p + k, K, lane);

        v16bf b0 = load_tile_bf16(b0p + k, K, lane);
        acc[0][0] = wmma_bf16(a0, b0, acc[0][0]);
        acc[1][0] = wmma_bf16(a1, b0, acc[1][0]);

        v16bf b1 = load_tile_bf16(b1p + k, K, lane);
        acc[0][1] = wmma_bf16(a0, b1, acc[0][1]);
        acc[1][1] = wmma_bf16(a1, b1, acc[1][1]);

        v16bf b2 = load_tile_bf16(b2p + k, K, lane);
        acc[0][2] = wmma_bf16(a0, b2, acc[0][2]);
        acc[1][2] = wmma_bf16(a1, b2, acc[1][2]);

        v16bf b3 = load_tile_bf16(b3p + k, K, lane);
        acc[0][3] = wmma_bf16(a0, b3, acc[0][3]);
        acc[1][3] = wmma_bf16(a1, b3, acc[1][3]);
    }

    const int rl = 8 * (lane >> 4);
    const int cl = lane & 15;
#pragma unroll
    for (int m = 0; m < 2; ++m)
#pragma unroll
        for (int n = 0; n < 4; ++n)
#pragma unroll
            for (int i = 0; i < 8; ++i)
                C[(size_t)(row0 + m * 16 + rl + i) * N + (col0 + n * 16 + cl)] =
                    acc[m][n][i];
}

// ---------------------------------------------------------------------------
// RoPE:  out = x*cos + rotate_half(x)*sin   (per position s, per head dim d)
// ---------------------------------------------------------------------------
__global__ void __launch_bounds__(256) k_rope(const float* __restrict__ x,
                                              const float* __restrict__ cosp,
                                              const float* __restrict__ sinp,
                                              __bf16* __restrict__ out) {
    int idx = blockIdx.x * 256 + threadIdx.x;
    if (idx >= SEQ * EMB) return;
    int s = idx / EMB;
    int e = idx - s * EMB;
    int d = e & (HD - 1);
    float xv = x[idx];
    float rot = (d < HD / 2) ? -x[idx + HD / 2] : x[idx - HD / 2];
    float r = xv * cosp[s * HD + d] + rot * sinp[s * HD + d];
    out[idx] = (__bf16)r;
}

// ---------------------------------------------------------------------------
// Fused flash attention over one (head, 64-row query block).
// V tile staged via GLOBAL_LOAD_ASYNC_TO_LDS_B128 (ASYNCcnt), then
// transposed in LDS into the WMMA B-operand layout.
// Also emits per-64x64-block score max (for gate_target).
// ---------------------------------------------------------------------------
__global__ void __launch_bounds__(256) k_attn(const __bf16* __restrict__ q,
                                              const __bf16* __restrict__ k,
                                              const __bf16* __restrict__ v,
                                              __bf16* __restrict__ ctx,
                                              float* __restrict__ blkmax) {
    const int h   = blockIdx.x;  // head
    const int qi  = blockIdx.y;  // query block
    const int tid = threadIdx.x;
    const int lane = tid & 31;
    const int w    = tid >> 5;

    __shared__ float sO[BLK][HD];                       // 32 KB accumulator
    __shared__ __align__(16) float  sS[BLK][72];        // 18 KB scaled scores
    __shared__ __align__(16) __bf16 sP[BLK][72];        //  9 KB exp(P) bf16
    __shared__ __align__(16) __bf16 sVraw[BLK][HD];     // 16 KB async-staged V
    __shared__ __align__(16) __bf16 sVt[HD][72];        // 18 KB V^T tile
    __shared__ float s_m[BLK], s_l[BLK], s_rmax[BLK];

    for (int i = tid; i < BLK * HD; i += 256) sO[i >> 7][i & (HD - 1)] = 0.f;
    if (tid < BLK) { s_m[tid] = -1e30f; s_l[tid] = 0.f; }
    __syncthreads();

    const __bf16* qbase = q + (size_t)(qi * BLK) * EMB + h * HD;
    const float scale = 0.08838834764831845f; // 1/sqrt(128)

    for (int j = 0; j <= qi; ++j) {
        // --- async-stage raw V block into LDS (64 x 128 bf16, 1024 x b128) ---
        const __bf16* vbase = v + (size_t)(j * BLK) * EMB + h * HD;
#pragma unroll
        for (int i = 0; i < 4; ++i) {
            int c   = tid + 256 * i;       // 0..1023 chunk id
            int key = c >> 4;              // 16 x b128 per key row
            int off = (c & 15) * 8;        // element offset within row
            async_copy_b128(&sVraw[key][off], vbase + (size_t)key * EMB + off);
        }
        wait_asynccnt0();
        __syncthreads();

        // --- transpose in LDS: sVt[d][key] = sVraw[key][d] ---
        for (int i = tid; i < BLK * HD; i += 256) {
            int key = i >> 7, d = i & (HD - 1);
            sVt[d][key] = sVraw[key][d];
        }
        __syncthreads();

        // --- S = scale * Q K^T, causal-masked, into sS ---
        {
            const __bf16* kbase = k + (size_t)(j * BLK) * EMB + h * HD;
            const int mt  = w >> 1;       // 0..3 query sub-tile
            const int ntb = (w & 1) * 2;  // 0 or 2 key sub-tile base
            const __bf16* aT = qbase + (size_t)(mt * 16) * EMB;
            v16bf a0 = load_tile_bf16(aT + 0,  EMB, lane);
            v16bf a1 = load_tile_bf16(aT + 32, EMB, lane);
            v16bf a2 = load_tile_bf16(aT + 64, EMB, lane);
            v16bf a3 = load_tile_bf16(aT + 96, EMB, lane);
#pragma unroll
            for (int t = 0; t < 2; ++t) {
                const int nt = ntb + t;
                const __bf16* bT = kbase + (size_t)(nt * 16) * EMB;
                v8f c = {};
                c = wmma_bf16(a0, load_tile_bf16(bT + 0,  EMB, lane), c);
                c = wmma_bf16(a1, load_tile_bf16(bT + 32, EMB, lane), c);
                c = wmma_bf16(a2, load_tile_bf16(bT + 64, EMB, lane), c);
                c = wmma_bf16(a3, load_tile_bf16(bT + 96, EMB, lane), c);
                const int rl = mt * 16 + 8 * (lane >> 4);
                const int cl = nt * 16 + (lane & 15);
#pragma unroll
                for (int i = 0; i < 8; ++i) {
                    float vsc = c[i] * scale;
                    if (j == qi && cl > rl + i) vsc = -1e30f;  // causal
                    sS[rl + i][cl] = vsc;
                }
            }
        }
        __syncthreads();

        // --- online softmax row update (one thread per query row) ---
        if (tid < BLK) {
            float tm = -1e30f;
            for (int n = 0; n < BLK; ++n) tm = fmaxf(tm, sS[tid][n]);
            s_rmax[tid] = tm;
            float mold = s_m[tid];
            float mnew = fmaxf(mold, tm);
            float corr = __expf(mold - mnew);
            float lsum = 0.f;
            for (int n = 0; n < BLK; ++n) {
                float e = __expf(sS[tid][n] - mnew);
                lsum += e;
                sP[tid][n] = (__bf16)e;
            }
            s_l[tid] = s_l[tid] * corr + lsum;
            s_m[tid] = mnew;
            for (int d = 0; d < HD; ++d) sO[tid][d] *= corr;
        }
        __syncthreads();

        if (tid == 0) {
            float bm = -1e30f;
            for (int r = 0; r < BLK; ++r) bm = fmaxf(bm, s_rmax[r]);
            blkmax[((size_t)h * NBK + qi) * NBK + j] = bm;
        }

        // --- O += P @ V  (wave owns 4 disjoint 16x16 output tiles) ---
        {
            const int qt  = w >> 1;        // 0..3
            const int dtb = (w & 1) * 4;   // 0 or 4
            v16bf a0 = load_tile_bf16(&sP[qt * 16][0],  72, lane);
            v16bf a1 = load_tile_bf16(&sP[qt * 16][32], 72, lane);
#pragma unroll
            for (int t = 0; t < 4; ++t) {
                const int dt = dtb + t;
                const int rl = qt * 16 + 8 * (lane >> 4);
                const int cl = dt * 16 + (lane & 15);
                v8f c;
#pragma unroll
                for (int i = 0; i < 8; ++i) c[i] = sO[rl + i][cl];
                c = wmma_bf16(a0, load_tile_bf16(&sVt[dt * 16][0],  72, lane), c);
                c = wmma_bf16(a1, load_tile_bf16(&sVt[dt * 16][32], 72, lane), c);
#pragma unroll
                for (int i = 0; i < 8; ++i) sO[rl + i][cl] = c[i];
            }
        }
        __syncthreads();
    }

    // --- normalize and write ctx (bf16, natural [S][E] layout) ---
    __bf16* cb = ctx + (size_t)(qi * BLK) * EMB + h * HD;
    for (int i = tid; i < BLK * HD; i += 256) {
        int r = i >> 7, d = i & (HD - 1);
        cb[(size_t)r * EMB + d] = (__bf16)(sO[r][d] / s_l[r]);
    }
}

// ---------------------------------------------------------------------------
// Per-(head, block) mean of 64 rows of a [S][E] bf16 tensor -> [H][NB][D] f32
// ---------------------------------------------------------------------------
__global__ void __launch_bounds__(128) k_block_mean(const __bf16* __restrict__ x,
                                                    float* __restrict__ out) {
    const int h = blockIdx.x, n = blockIdx.y, d = threadIdx.x;
    const __bf16* p = x + (size_t)(n * BLK) * EMB + h * HD + d;
    float s = 0.f;
    for (int r = 0; r < BLK; ++r) s += (float)p[(size_t)r * EMB];
    out[((size_t)h * NBK + n) * HD + d] = s * (1.f / 64.f);
}

// ---------------------------------------------------------------------------
// gate_pred: per head, project block means through Wg, score, causal softmax
// ---------------------------------------------------------------------------
__global__ void __launch_bounds__(256) k_gate_pred(const float* __restrict__ qb,
                                                   const float* __restrict__ kb,
                                                   const float* __restrict__ Wgq,
                                                   const float* __restrict__ Wgk,
                                                   float* __restrict__ out) {
    const int h = blockIdx.x, tid = threadIdx.x;
    __shared__ float gq[NBK * GD], gk[NBK * GD], gl[NBK * NBK];
    __shared__ float red[256];

    const float* qbh = qb + (size_t)h * NBK * HD;
    const float* kbh = kb + (size_t)h * NBK * HD;
    for (int t = tid; t < NBK * GD; t += 256) {
        int n = t >> 6, g = t & 63;
        float sq = 0.f, sk = 0.f;
        for (int d = 0; d < HD; ++d) {
            sq += qbh[n * HD + d] * Wgq[g * HD + d];
            sk += kbh[n * HD + d] * Wgk[g * HD + d];
        }
        gq[t] = sq; gk[t] = sk;
    }
    __syncthreads();

    for (int t = tid; t < NBK * NBK; t += 256) {
        int qn = t >> 5, kn = t & 31;
        float vv = -1e30f;
        if (kn <= qn) {
            float s = 0.f;
            for (int g = 0; g < GD; ++g) s += gq[qn * GD + g] * gk[kn * GD + g];
            vv = s * 0.125f;  // 1/sqrt(64)
        }
        gl[t] = vv;
    }
    __syncthreads();

    float lm = -1e30f;
    for (int t = tid; t < NBK * NBK; t += 256) lm = fmaxf(lm, gl[t]);
    red[tid] = lm; __syncthreads();
    for (int s = 128; s > 0; s >>= 1) { if (tid < s) red[tid] = fmaxf(red[tid], red[tid + s]); __syncthreads(); }
    const float mx = red[0]; __syncthreads();

    float ls = 0.f;
    for (int t = tid; t < NBK * NBK; t += 256) {
        float e = (gl[t] <= -1e29f) ? 0.f : __expf(gl[t] - mx);
        gl[t] = e; ls += e;
    }
    red[tid] = ls; __syncthreads();
    for (int s = 128; s > 0; s >>= 1) { if (tid < s) red[tid] += red[tid + s]; __syncthreads(); }
    const float inv = 1.f / red[0];
    for (int t = tid; t < NBK * NBK; t += 256) out[(size_t)h * NBK * NBK + t] = gl[t] * inv;
}

// ---------------------------------------------------------------------------
// gate_target: per head, clamp block max, temp softmax over causal blocks
// ---------------------------------------------------------------------------
__global__ void __launch_bounds__(256) k_gate_target(const float* __restrict__ bm,
                                                     float* __restrict__ out) {
    const int h = blockIdx.x, tid = threadIdx.x;
    __shared__ float gl[NBK * NBK];
    __shared__ float red[256];

    for (int t = tid; t < NBK * NBK; t += 256) {
        int qn = t >> 5, kn = t & 31;
        float x = -1e30f;
        if (kn <= qn) {
            float b = bm[((size_t)h * NBK + qn) * NBK + kn];
            b = fminf(fmaxf(b, -50.f), 50.f);
            x = b * 0.5f;  // / GATE_TEMP
        }
        gl[t] = x;
    }
    __syncthreads();

    float lm = -1e30f;
    for (int t = tid; t < NBK * NBK; t += 256) lm = fmaxf(lm, gl[t]);
    red[tid] = lm; __syncthreads();
    for (int s = 128; s > 0; s >>= 1) { if (tid < s) red[tid] = fmaxf(red[tid], red[tid + s]); __syncthreads(); }
    const float mx = red[0]; __syncthreads();

    float ls = 0.f;
    for (int t = tid; t < NBK * NBK; t += 256) {
        float e = (gl[t] <= -1e29f) ? 0.f : __expf(gl[t] - mx);
        gl[t] = e; ls += e;
    }
    red[tid] = ls; __syncthreads();
    for (int s = 128; s > 0; s >>= 1) { if (tid < s) red[tid] += red[tid + s]; __syncthreads(); }
    const float inv = 1.f / red[0];
    for (int t = tid; t < NBK * NBK; t += 256) out[(size_t)h * NBK * NBK + t] = gl[t] * inv;
}

// ---------------------------------------------------------------------------
// Host launcher
// ---------------------------------------------------------------------------
extern "C" void kernel_launch(void* const* d_in, const int* in_sizes, int n_in,
                              void* d_out, int out_size, void* d_ws, size_t ws_size,
                              hipStream_t stream) {
    (void)in_sizes; (void)n_in; (void)out_size; (void)ws_size;

    const float* hs   = (const float*)d_in[0];
    const float* cosp = (const float*)d_in[1];
    const float* sinp = (const float*)d_in[2];
    const float* Wq   = (const float*)d_in[3];
    const float* Wk   = (const float*)d_in[4];
    const float* Wv   = (const float*)d_in[5];
    const float* Wo   = (const float*)d_in[6];
    const float* Wgq  = (const float*)d_in[7];
    const float* Wgk  = (const float*)d_in[8];

    const size_t SE = (size_t)SEQ * EMB;
    const size_t EE = (size_t)EMB * EMB;

    char* ws = (char*)d_ws;
    size_t off = 0;
    auto take = [&](size_t bytes) -> char* {
        char* p = ws + off;
        off += (bytes + 255) & ~(size_t)255;
        return p;
    };

    __bf16* hs_b  = (__bf16*)take(SE * 2);
    __bf16* wq_b  = (__bf16*)take(EE * 2);
    __bf16* wk_b  = (__bf16*)take(EE * 2);
    __bf16* wv_b  = (__bf16*)take(EE * 2);
    __bf16* wo_b  = (__bf16*)take(EE * 2);
    float*  q_f   = (float*)take(SE * 4);
    float*  k_f   = (float*)take(SE * 4);
    float*  v_f   = (float*)take(SE * 4);
    __bf16* q_b   = (__bf16*)take(SE * 2);
    __bf16* k_b   = (__bf16*)take(SE * 2);
    __bf16* v_b   = (__bf16*)take(SE * 2);
    __bf16* ctx_b = (__bf16*)take(SE * 2);
    float*  bmax  = (float*)take((size_t)NH * NBK * NBK * 4);
    float*  qbm   = (float*)take((size_t)NH * NBK * HD * 4);
    float*  kbm   = (float*)take((size_t)NH * NBK * HD * 4);

    float* out_attn = (float*)d_out;
    float* out_gp   = out_attn + SE;
    float* out_gt   = out_gp + (size_t)NH * NBK * NBK;

    // 1) f32 -> bf16 conversions
    k_f32_to_bf16<<<(int)((SE + 255) / 256), 256, 0, stream>>>(hs, hs_b, (int)SE);
    k_f32_to_bf16<<<(int)((EE + 255) / 256), 256, 0, stream>>>(Wq, wq_b, (int)EE);
    k_f32_to_bf16<<<(int)((EE + 255) / 256), 256, 0, stream>>>(Wk, wk_b, (int)EE);
    k_f32_to_bf16<<<(int)((EE + 255) / 256), 256, 0, stream>>>(Wv, wv_b, (int)EE);
    k_f32_to_bf16<<<(int)((EE + 255) / 256), 256, 0, stream>>>(Wo, wo_b, (int)EE);

    // 2) Q/K/V projections (WMMA GEMM, A * W^T); macro tile 64 x 256
    dim3 ggrid(SEQ / 64, EMB / 256);
    k_gemm_bf16_nt<<<ggrid, 256, 0, stream>>>(hs_b, wq_b, q_f, SEQ, EMB, EMB);
    k_gemm_bf16_nt<<<ggrid, 256, 0, stream>>>(hs_b, wk_b, k_f, SEQ, EMB, EMB);
    k_gemm_bf16_nt<<<ggrid, 256, 0, stream>>>(hs_b, wv_b, v_f, SEQ, EMB, EMB);

    // 3) RoPE on q/k -> bf16, v -> bf16
    k_rope<<<(int)(SE / 256), 256, 0, stream>>>(q_f, cosp, sinp, q_b);
    k_rope<<<(int)(SE / 256), 256, 0, stream>>>(k_f, cosp, sinp, k_b);
    k_f32_to_bf16<<<(int)((SE + 255) / 256), 256, 0, stream>>>(v_f, v_b, (int)SE);

    // 4) fused flash attention + block-max extraction
    k_attn<<<dim3(NH, NBK), 256, 0, stream>>>(q_b, k_b, v_b, ctx_b, bmax);

    // 5) output projection -> d_out
    k_gemm_bf16_nt<<<ggrid, 256, 0, stream>>>(ctx_b, wo_b, out_attn, SEQ, EMB, EMB);

    // 6) gate heads
    k_block_mean<<<dim3(NH, NBK), 128, 0, stream>>>(q_b, qbm);
    k_block_mean<<<dim3(NH, NBK), 128, 0, stream>>>(k_b, kbm);
    k_gate_pred<<<NH, 256, 0, stream>>>(qbm, kbm, Wgq, Wgk, out_gp);
    k_gate_target<<<NH, 256, 0, stream>>>(bmax, out_gt);
}
